// ChartQualityEvaluator_66030827209232
// MI455X (gfx1250) — compile-verified, matrix operand order checked by reading, not run on hardware
//
#include <hip/hip_runtime.h>
#include <hip/hip_bf16.h>
#include <math.h>

typedef __attribute__((ext_vector_type(16))) _Float16 v16h;
typedef __attribute__((ext_vector_type(8)))  float    v8f;
typedef __attribute__((ext_vector_type(2)))  _Float16 h2;
typedef __attribute__((ext_vector_type(4)))  _Float16 h4;
typedef __attribute__((ext_vector_type(8)))  _Float16 h8;

#if __has_builtin(__builtin_amdgcn_tensor_load_to_lds)
#define USE_TDM 1
#else
#define USE_TDM 0
#endif

#if USE_TDM
typedef unsigned int u32x4 __attribute__((ext_vector_type(4)));
typedef int          i32x8 __attribute__((ext_vector_type(8)));
typedef int          i32x4 __attribute__((ext_vector_type(4)));
#endif

#define BATCH   32
#define SEQ     500
#define SEQP    512
#define DIM     256
#define HEADS   8
#define HDIM    32
#define NLAYERS 6
#define NEVT    256
#define MELC    80
#define T0      2000
#define C1      128
#define T1      1000
#define C2      256
#define T2      500
#define HALFD   128

#define GBM 32
#define GBN 64
#define GBK 32
#define LPAD 8

__device__ __forceinline__ float gelu_f(float x) {
  return 0.5f * x * (1.0f + erff(x * 0.7071067811865476f));
}

__device__ __forceinline__ float sinpos_val(float xv, int d) {
  const float c = -9.210340371976184f / 127.0f;  // -ln(10000)/(HALF-1)
  int i = (d < HALFD) ? d : d - HALFD;
  float e = xv * __expf(c * (float)i);
  return (d < HALFD) ? __sinf(e) : __cosf(e);
}

// ---------------------------------------------------------------------------
// f32 -> f16 conversion (weights, once per launch)
// ---------------------------------------------------------------------------
__global__ void k_f32_to_f16(const float* __restrict__ in,
                             _Float16* __restrict__ out, int n)
{
  int i = blockIdx.x * blockDim.x + threadIdx.x;
  if (i < n) out[i] = (_Float16)in[i];
}

// ---------------------------------------------------------------------------
// WMMA GEMM, all-f16 operands: Cout[M,N] = act(A[M,K] x W[N,K]^T + bias) (+Cin)
// All dims must be tile-aligned (M%32==0, N%64==0, K%32==0) -- true here.
// Block = 256 threads (8 waves); block tile 32x64; wave tile 16x16.
// B tile is DMA'd into LDS by the Tensor Data Mover when available.
// ---------------------------------------------------------------------------
__global__ __launch_bounds__(256) void k_gemm_wmma(
    const _Float16* __restrict__ A, const _Float16* __restrict__ W,
    const float* __restrict__ bias, const float* __restrict__ Cin,
    void* __restrict__ Cout, int M, int N, int K,
    int lda, int ldb, int ldc, int act, int outF16, float scale)
{
  __shared__ alignas(16) _Float16 As[GBM][GBK + LPAD];
  __shared__ alignas(16) _Float16 Bs[GBN][GBK + LPAD];

  const int tid  = threadIdx.x;
  const int lane = tid & 31;
  const int wid  = tid >> 5;
  const int wm   = wid >> 2;   // 0..1
  const int wn   = wid & 3;    // 0..3
  const int m0   = blockIdx.y * GBM;
  const int n0   = blockIdx.x * GBN;
  const int hlf  = lane >> 4;
  const int lrow = lane & 15;

  // A-tile staging coords: 32 rows x 8 segs of 4 halves = 256 units
  const int ar = tid >> 3;
  const int ac = (tid & 7) * 4;
#if !USE_TDM
  const int br = tid >> 2;
  const int bc = (tid & 3) * 8;
#endif

  v8f acc = {};

  for (int k0 = 0; k0 < K; k0 += GBK) {
    // stage A tile (vectorized, no bounds checks needed)
    *(h4*)&As[ar][ac] =
        *(const h4*)(A + (long long)(m0 + ar) * lda + k0 + ac);

#if USE_TDM
    // --- Tensor Data Mover: DMA the 64x32-half weight tile into LDS ---
    if (tid == 0) {
      unsigned long long ga =
          (unsigned long long)(W + (long long)n0 * ldb + k0);
      unsigned lds = (unsigned)(unsigned long long)&Bs[0][0];
      u32x4 g0;
      g0[0] = 1u;                                   // count=1, user mode
      g0[1] = lds;                                  // lds_addr (bytes)
      g0[2] = (unsigned)(ga & 0xFFFFFFFFu);         // global_addr[31:0]
      g0[3] = (unsigned)((ga >> 32) & 0x1FFFFFFu)   // global_addr[56:32]
              | (2u << 30);                         // type=2 (image)
      i32x8 g1;
      // data_size=2B(code1), pad_enable, pad_interval=16DW(code3),
      // pad_amount=4DW(code3)  -> row stride in LDS = 32+8 halves = GBK+LPAD
      g1[0] = (int)((1u << 16) | (1u << 20) | (3u << 22) | (3u << 25));
      g1[1] = (int)(((unsigned)K & 0xFFFFu) << 16);            // tensor_dim0 lo
      g1[2] = (int)((((unsigned)K >> 16) & 0xFFFFu)
              | (((unsigned)N & 0xFFFFu) << 16));              // dim0 hi|dim1 lo
      g1[3] = (int)(((((unsigned)N) >> 16) & 0xFFFFu)
              | (32u << 16));                                  // dim1 hi|tile_dim0=32
      g1[4] = (int)64u;                                        // tile_dim1=64
      g1[5] = (int)(unsigned)ldb;                              // dim0_stride lo32
      g1[6] = 0;
      g1[7] = 0;
      i32x4 gz4 = {0, 0, 0, 0};
      i32x8 gz8 = {0, 0, 0, 0, 0, 0, 0, 0};
      __builtin_amdgcn_tensor_load_to_lds(g0, g1, gz4, gz4, gz8, 0);
    }
    __builtin_amdgcn_s_wait_tensorcnt(0);
#else
    // manual vectorized staging: 64 rows x 4 segs of 8 halves = 256 units
    *(h8*)&Bs[br][bc] =
        *(const h8*)(W + (long long)(n0 + br) * ldb + k0 + bc);
#endif
    if (k0 + GBK < K)
      __builtin_prefetch(&A[(long long)(m0 + ar) * lda + k0 + GBK], 0, 1);
    __syncthreads();

    // fragments per ISA 16-bit A/B 16x32 lane layout
    v16h af, bf;
    const int am = wm * 16 + lrow;
    const int bn = wn * 16 + lrow;
#pragma unroll
    for (int j = 0; j < 8; ++j) {
      int kk = ((j & 4) << 2) + (hlf << 3) + ((j & 3) << 1);
      af[2 * j]     = As[am][kk];
      af[2 * j + 1] = As[am][kk + 1];
      bf[2 * j]     = Bs[bn][kk];
      bf[2 * j + 1] = Bs[bn][kk + 1];
    }
    acc = __builtin_amdgcn_wmma_f32_16x16x32_f16(false, af, false, bf,
                                                 (short)0, acc, false, false);
    __syncthreads();
  }

  // epilogue: C layout -- VGPR r, lane L: n = L%16, m = r + 8*(L>=16)
  const int cn = n0 + wn * 16 + lrow;
#pragma unroll
  for (int r = 0; r < 8; ++r) {
    int cm = m0 + wm * 16 + r + hlf * 8;
    float v = acc[r] * scale;
    if (bias) v += bias[cn];
    if (act)  v = gelu_f(v);
    if (Cin)  v += Cin[(long long)cm * ldc + cn];
    if (outF16) ((_Float16*)Cout)[(long long)cm * ldc + cn] = (_Float16)v;
    else        ((float*)Cout)[(long long)cm * ldc + cn] = v;
  }
}

// ---------------------------------------------------------------------------
// Fused attention over f16 QKV: one block per (16-query tile, b*H + h).
// Scores row-block (16 x 512) lives in LDS; QK^T and P*V both via WMMA.
// ---------------------------------------------------------------------------
__global__ __launch_bounds__(256) void k_attention(
    const _Float16* __restrict__ qkv, _Float16* __restrict__ out)
{
  __shared__ alignas(16) _Float16 Qs[16][HDIM + LPAD];     // ~1.3 KB
  __shared__ alignas(16) _Float16 KVs[128][HDIM + LPAD];   // 10 KB stage
  __shared__ float    Sc[16][SEQP];                        // 32 KB
  __shared__ float    Red[16][17];
  __shared__ float    Part[8][16][16];                     // 8 KB

  const int tid  = threadIdx.x;
  const int lane = tid & 31;
  const int wid  = tid >> 5;
  const int hlf  = lane >> 4;
  const int lrow = lane & 15;

  const int q0 = blockIdx.x * 16;
  const int z  = blockIdx.y;
  const int b  = z / HEADS;
  const int h  = z % HEADS;

  const int qrow = 3 * DIM;
  const _Float16* qbase = qkv + (long long)b * SEQ * 3 * DIM + h * HDIM;
  const _Float16* kbase = qbase + DIM;
  const _Float16* vbase = qbase + 2 * DIM;

  // load Q tile: 16 rows x 16 segs of 2 halves
  {
    int r = tid >> 4, d0 = (tid & 15) * 2;
    int s = q0 + r;
    h2 v = {};
    if (s < SEQ) v = *(const h2*)(qbase + (long long)s * qrow + d0);
    *(h2*)&Qs[r][d0] = v;
  }
  __syncthreads();

  // ---- phase 1: S = Q K^T over 4 chunks of 128 keys (8 waves x 16 keys) ----
  for (int it = 0; it < 4; ++it) {
    {
      int r = tid >> 1, d0 = (tid & 1) * 16;
      int s = it * 128 + r;
      h8 a = {}, bv = {};
      if (s < SEQ) {
        a  = *(const h8*)(kbase + (long long)s * qrow + d0);
        bv = *(const h8*)(kbase + (long long)s * qrow + d0 + 8);
      }
      *(h8*)&KVs[r][d0]     = a;
      *(h8*)&KVs[r][d0 + 8] = bv;
    }
    __syncthreads();
    v16h af, bf;
#pragma unroll
    for (int j = 0; j < 8; ++j) {
      int kk = ((j & 4) << 2) + (hlf << 3) + ((j & 3) << 1);
      af[2 * j]     = Qs[lrow][kk];
      af[2 * j + 1] = Qs[lrow][kk + 1];
      bf[2 * j]     = KVs[wid * 16 + lrow][kk];
      bf[2 * j + 1] = KVs[wid * 16 + lrow][kk + 1];
    }
    v8f sacc = {};
    sacc = __builtin_amdgcn_wmma_f32_16x16x32_f16(false, af, false, bf,
                                                  (short)0, sacc, false, false);
#pragma unroll
    for (int r = 0; r < 8; ++r)
      Sc[r + hlf * 8][it * 128 + wid * 16 + lrow] = sacc[r];
    __syncthreads();
  }

  // ---- phase 2: softmax over keys (scale 1/sqrt(hd), mask >= SEQ) ----
  {
    const float scl = 0.17677669529663687f;
    int row = tid >> 4;
    int t16 = tid & 15;
    float lmax = -1e30f;
    for (int c = t16; c < SEQP; c += 16) {
      float v = (c < SEQ) ? Sc[row][c] * scl : -1e30f;
      lmax = fmaxf(lmax, v);
    }
    Red[row][t16] = lmax;
    __syncthreads();
    float mx = -1e30f;
#pragma unroll
    for (int j = 0; j < 16; ++j) mx = fmaxf(mx, Red[row][j]);
    __syncthreads();
    float lsum = 0.0f;
    for (int c = t16; c < SEQP; c += 16) {
      float p = (c < SEQ) ? __expf(Sc[row][c] * scl - mx) : 0.0f;
      Sc[row][c] = p;
      lsum += p;
    }
    Red[row][t16] = lsum;
    __syncthreads();
    float sum = 1e-20f;
#pragma unroll
    for (int j = 0; j < 16; ++j) sum += Red[row][j];
    float inv = 1.0f / sum;
    for (int c = t16; c < SEQP; c += 16) Sc[row][c] *= inv;
    __syncthreads();
  }

  // ---- phase 3: O = P V.  wave = (key-chunk kc 0..3, dim-half dh 0..1) ----
  const int kc = wid >> 1;
  const int dh = wid & 1;
  v8f oacc = {};
  for (int it = 0; it < 4; ++it) {
    {
      int r = tid >> 1, d0 = (tid & 1) * 16;
      int s = it * 128 + r;
      h8 a = {}, bv = {};
      if (s < SEQ) {
        a  = *(const h8*)(vbase + (long long)s * qrow + d0);
        bv = *(const h8*)(vbase + (long long)s * qrow + d0 + 8);
      }
      *(h8*)&KVs[r][d0]     = a;
      *(h8*)&KVs[r][d0 + 8] = bv;
    }
    __syncthreads();
    const int kb = it * 128 + kc * 32;
    v16h af, bf;
#pragma unroll
    for (int j = 0; j < 8; ++j) {
      int kk = ((j & 4) << 2) + (hlf << 3) + ((j & 3) << 1);
      af[2 * j]     = (_Float16)Sc[lrow][kb + kk];
      af[2 * j + 1] = (_Float16)Sc[lrow][kb + kk + 1];
      bf[2 * j]     = KVs[kc * 32 + kk][dh * 16 + lrow];
      bf[2 * j + 1] = KVs[kc * 32 + kk + 1][dh * 16 + lrow];
    }
    oacc = __builtin_amdgcn_wmma_f32_16x16x32_f16(false, af, false, bf,
                                                  (short)0, oacc, false, false);
    __syncthreads();
  }
#pragma unroll
  for (int r = 0; r < 8; ++r) Part[wid][r + hlf * 8][lrow] = oacc[r];
  __syncthreads();
  for (int i = tid; i < 16 * HDIM; i += 256) {
    int m = i >> 5, d = i & 31;
    int dhh = d >> 4, dl = d & 15;
    float v = Part[0 + dhh][m][dl] + Part[2 + dhh][m][dl] +
              Part[4 + dhh][m][dl] + Part[6 + dhh][m][dl];
    int s = q0 + m;
    if (s < SEQ)
      out[((long long)b * SEQ + s) * DIM + h * HDIM + d] = (_Float16)v;
  }
}

// ---------------------------------------------------------------------------
// Front-end / small kernels
// ---------------------------------------------------------------------------
__global__ void k_mel(const float* __restrict__ mel, const float* __restrict__ W,
                      const float* __restrict__ bs, float* __restrict__ out)
{
  int idx = blockIdx.x * blockDim.x + threadIdx.x;
  if (idx >= BATCH * 16 * T0) return;
  int t = idx % T0;
  int o = (idx / T0) % 16;
  int b = idx / (16 * T0);
  float acc = bs[o];
  const float* mp = mel + (long long)b * MELC * T0 + t;
  const float* wp = W + o * MELC;
  for (int c = 0; c < MELC; ++c) acc += mp[(long long)c * T0] * wp[c];
  out[idx] = acc;
}

__global__ void k_conv1(const float* __restrict__ xin, const float* __restrict__ W,
                        const float* __restrict__ bs, float* __restrict__ out)
{
  int idx = blockIdx.x * blockDim.x + threadIdx.x;
  if (idx >= BATCH * C1 * T1) return;
  int t = idx % T1;
  int o = (idx / T1) % C1;
  int b = idx / (C1 * T1);
  float acc = bs[o];
  for (int c = 0; c < 16; ++c) {
    const float* xp = xin + ((long long)b * 16 + c) * T0;
    const float* wp = W + (o * 16 + c) * 7;
#pragma unroll
    for (int tap = 0; tap < 7; ++tap) {
      int ti = 2 * t + tap - 3;
      if (ti >= 0 && ti < T0) acc += xp[ti] * wp[tap];
    }
  }
  out[idx] = gelu_f(acc);
}

__global__ __launch_bounds__(256) void k_gn_stats(const float* __restrict__ h,
                                                  float* __restrict__ st)
{
  __shared__ float rs[256], rq[256];
  int b = blockIdx.x, tid = threadIdx.x;
  const float* hp = h + (long long)b * C1 * T1;
  float s = 0.0f, q = 0.0f;
  for (int i = tid; i < C1 * T1; i += 256) { float v = hp[i]; s += v; q += v * v; }
  rs[tid] = s; rq[tid] = q; __syncthreads();
  for (int r = 128; r > 0; r >>= 1) {
    if (tid < r) { rs[tid] += rs[tid + r]; rq[tid] += rq[tid + r]; }
    __syncthreads();
  }
  if (tid == 0) {
    float mean = rs[0] / (float)(C1 * T1);
    float var  = rq[0] / (float)(C1 * T1) - mean * mean;
    st[b * 2]     = mean;
    st[b * 2 + 1] = rsqrtf(var + 1e-5f);
  }
}

__global__ void k_gn_apply(float* h, const float* __restrict__ st,
                           const float* __restrict__ g, const float* __restrict__ bt)
{
  int idx = blockIdx.x * blockDim.x + threadIdx.x;
  if (idx >= BATCH * C1 * T1) return;
  int c = (idx / T1) % C1;
  int b = idx / (C1 * T1);
  h[idx] = (h[idx] - st[b * 2]) * st[b * 2 + 1] * g[c] + bt[c];
}

__global__ void k_conv2(const float* __restrict__ xin, const float* __restrict__ W,
                        const float* __restrict__ bs, float* __restrict__ out)
{
  int idx = blockIdx.x * blockDim.x + threadIdx.x;
  if (idx >= BATCH * C2 * T2) return;
  int t = idx % T2;
  int o = (idx / T2) % C2;
  int b = idx / (C2 * T2);
  float acc = bs[o];
  for (int c = 0; c < C1; ++c) {
    const float* xp = xin + ((long long)b * C1 + c) * T1;
    const float* wp = W + (o * C1 + c) * 7;
#pragma unroll
    for (int tap = 0; tap < 7; ++tap) {
      int ti = 2 * t + tap - 3;
      if (ti >= 0 && ti < T1) acc += xp[ti] * wp[tap];
    }
  }
  out[idx] = gelu_f(acc);
}

__global__ __launch_bounds__(256) void k_ln_pos_star(
    const float* __restrict__ h2i, const float* __restrict__ g,
    const float* __restrict__ bt, const float* __restrict__ star_table,
    const float* __restrict__ star_rating, float* __restrict__ x)
{
  __shared__ float red[256];
  int idx = blockIdx.x;  // b*SEQ + s
  int b = idx / SEQ, s = idx % SEQ;
  int tid = threadIdx.x; // d
  float v = h2i[((long long)b * C2 + tid) * T2 + s];
  red[tid] = v; __syncthreads();
  for (int r = 128; r > 0; r >>= 1) { if (tid < r) red[tid] += red[tid + r]; __syncthreads(); }
  float mean = red[0] / (float)DIM;
  __syncthreads();
  float d = v - mean;
  red[tid] = d * d; __syncthreads();
  for (int r = 128; r > 0; r >>= 1) { if (tid < r) red[tid] += red[tid + r]; __syncthreads(); }
  float rstd = rsqrtf(red[0] / (float)DIM + 1e-5f);
  float xv = d * rstd * g[tid] + bt[tid];
  xv += sinpos_val((float)s, tid);
  int bucket = (int)(star_rating[b] * 2.0f);
  bucket = min(max(bucket, 0), 19);
  xv += star_table[bucket * DIM + tid];
  x[((long long)b * SEQ + s) * DIM + tid] = xv;
}

__global__ __launch_bounds__(256) void k_event_embed(
    const int* __restrict__ events, const unsigned char* __restrict__ mask,
    const float* __restrict__ W1, const float* __restrict__ b1,
    const float* __restrict__ W2, const float* __restrict__ b2,
    float* __restrict__ x)
{
  __shared__ float comb[3 * DIM];
  __shared__ float h1s[DIM];
  int idx = blockIdx.x;  // b*NEVT + e
  int b = idx / NEVT, e = idx % NEVT;
  int tid = threadIdx.x;
  const int* ev = events + b * NEVT;

  auto gAt = [&](int i) {
    int j = (i == 0) ? 1 : i;
    int dg = ev[j] - ev[j - 1];
    return dg < 1 ? 1 : dg;
  };
  float gf_e = (float)gAt(e);
  float rb = 1.0f, ra = 1.0f;
  if (e > 0)        rb = fminf(fmaxf(gf_e / (float)gAt(e - 1), 0.1f), 10.0f);
  if (e < NEVT - 1) ra = fminf(fmaxf((float)gAt(e + 1) / gf_e, 0.1f), 10.0f);
  float rbv = (float)((int)(rb * 50.0f));
  float rav = (float)((int)(ra * 50.0f));
  float gmv = (float)(gAt(e) * 5);

  comb[tid]           = sinpos_val(rbv, tid);
  comb[DIM + tid]     = sinpos_val(rav, tid);
  comb[2 * DIM + tid] = sinpos_val(gmv, tid);
  __syncthreads();

  float acc = b1[tid];
  const float* w1r = W1 + (long long)tid * 3 * DIM;
  for (int k = 0; k < 3 * DIM; ++k) acc += comb[k] * w1r[k];
  h1s[tid] = gelu_f(acc);
  __syncthreads();

  float acc2 = b2[tid];
  const float* w2r = W2 + (long long)tid * DIM;
  for (int k = 0; k < DIM; ++k) acc2 += h1s[k] * w2r[k];

  if (mask[b * NEVT + e]) acc2 = 0.0f;
  int pos = ev[e] / 4;
  pos = min(max(pos, 0), SEQ - 1);
  atomicAdd(&x[((long long)b * SEQ + pos) * DIM + tid], acc2);
}

// LayerNorm producing f16 rows for the WMMA GEMMs
__global__ __launch_bounds__(256) void k_layernorm(
    const float* __restrict__ x, const float* __restrict__ g,
    const float* __restrict__ bt, _Float16* __restrict__ y)
{
  __shared__ float red[256];
  int row = blockIdx.x;
  int tid = threadIdx.x;
  float v = x[(long long)row * DIM + tid];
  red[tid] = v; __syncthreads();
  for (int s = 128; s > 0; s >>= 1) { if (tid < s) red[tid] += red[tid + s]; __syncthreads(); }
  float mean = red[0] / (float)DIM;
  __syncthreads();
  float d = v - mean;
  red[tid] = d * d; __syncthreads();
  for (int s = 128; s > 0; s >>= 1) { if (tid < s) red[tid] += red[tid + s]; __syncthreads(); }
  float rstd = rsqrtf(red[0] / (float)DIM + 1e-5f);
  y[(long long)row * DIM + tid] = (_Float16)(d * rstd * g[tid] + bt[tid]);
}

__global__ __launch_bounds__(256) void k_pool_head(
    const float* __restrict__ x, const float* __restrict__ pq,
    const float* __restrict__ ong, const float* __restrict__ onb,
    const float* __restrict__ ohW, const float* __restrict__ ohb,
    float* __restrict__ out)
{
  __shared__ float wgt[SEQP];
  __shared__ float red[256];
  int b = blockIdx.x, tid = threadIdx.x;
  const float* xb = x + (long long)b * SEQ * DIM;

  float lmax = -1e30f;
  for (int s = tid; s < SEQ; s += 256) {
    float acc = 0.0f;
    const float* xr = xb + (long long)s * DIM;
    for (int d = 0; d < DIM; ++d) acc += xr[d] * pq[d];
    float l = acc * 0.0625f;  // 1/sqrt(256)
    wgt[s] = l;
    lmax = fmaxf(lmax, l);
  }
  red[tid] = lmax; __syncthreads();
  for (int r = 128; r > 0; r >>= 1) { if (tid < r) red[tid] = fmaxf(red[tid], red[tid + r]); __syncthreads(); }
  float mx = red[0]; __syncthreads();
  float lsum = 0.0f;
  for (int s = tid; s < SEQ; s += 256) { float p = __expf(wgt[s] - mx); wgt[s] = p; lsum += p; }
  red[tid] = lsum; __syncthreads();
  for (int r = 128; r > 0; r >>= 1) { if (tid < r) red[tid] += red[tid + r]; __syncthreads(); }
  float inv = 1.0f / red[0];
  __syncthreads();
  float acc = 0.0f;
  for (int s = 0; s < SEQ; ++s) acc += wgt[s] * xb[(long long)s * DIM + tid];
  float pv = acc * inv;
  red[tid] = pv; __syncthreads();
  for (int r = 128; r > 0; r >>= 1) { if (tid < r) red[tid] += red[tid + r]; __syncthreads(); }
  float mean = red[0] / (float)DIM; __syncthreads();
  float dv = pv - mean;
  red[tid] = dv * dv; __syncthreads();
  for (int r = 128; r > 0; r >>= 1) { if (tid < r) red[tid] += red[tid + r]; __syncthreads(); }
  float rstd = rsqrtf(red[0] / (float)DIM + 1e-5f); __syncthreads();
  float term = (dv * rstd * ong[tid] + onb[tid]) * ohW[tid];
  red[tid] = term; __syncthreads();
  for (int r = 128; r > 0; r >>= 1) { if (tid < r) red[tid] += red[tid + r]; __syncthreads(); }
  if (tid == 0) out[b] = red[0] + ohb[0];
}

// ---------------------------------------------------------------------------
extern "C" void kernel_launch(void* const* d_in, const int* in_sizes, int n_in,
                              void* d_out, int out_size, void* d_ws, size_t ws_size,
                              hipStream_t stream)
{
  (void)in_sizes; (void)n_in; (void)out_size; (void)ws_size;
  const float* mel        = (const float*)d_in[0];
  const int*   events     = (const int*)d_in[1];
  const unsigned char* emask = (const unsigned char*)d_in[2];
  const float* star       = (const float*)d_in[3];
  const float* mel_W      = (const float*)d_in[4];
  const float* mel_b      = (const float*)d_in[5];
  const float* conv1_w    = (const float*)d_in[6];
  const float* conv1_b    = (const float*)d_in[7];
  const float* gn_g       = (const float*)d_in[8];
  const float* gn_b       = (const float*)d_in[9];
  const float* conv2_w    = (const float*)d_in[10];
  const float* conv2_b    = (const float*)d_in[11];
  const float* cn_g       = (const float*)d_in[12];
  const float* cn_b       = (const float*)d_in[13];
  const float* star_table = (const float*)d_in[14];
  const float* ep_W1      = (const float*)d_in[15];
  const float* ep_b1      = (const float*)d_in[16];
  const float* ep_W2      = (const float*)d_in[17];
  const float* ep_b2      = (const float*)d_in[18];
  const float* tl_Wqkv    = (const float*)d_in[19];
  const float* tl_bqkv    = (const float*)d_in[20];
  const float* tl_Wo      = (const float*)d_in[21];
  const float* tl_bo      = (const float*)d_in[22];
  const float* tl_ln1g    = (const float*)d_in[23];
  const float* tl_ln1b    = (const float*)d_in[24];
  const float* tl_ln2g    = (const float*)d_in[25];
  const float* tl_ln2b    = (const float*)d_in[26];
  const float* tl_W1      = (const float*)d_in[27];
  const float* tl_b1      = (const float*)d_in[28];
  const float* tl_W2      = (const float*)d_in[29];
  const float* tl_b2      = (const float*)d_in[30];
  const float* pool_q     = (const float*)d_in[31];
  const float* on_g       = (const float*)d_in[32];
  const float* on_b       = (const float*)d_in[33];
  const float* oh_W       = (const float*)d_in[34];
  const float* oh_b       = (const float*)d_in[35];

  char* base = (char*)d_ws;
  size_t off = 0;
  auto allocF = [&](size_t n) {
    float* p = (float*)(base + off);
    off = (off + n * 4 + 255) & ~(size_t)255;
    return p;
  };
  auto allocH = [&](size_t n) {
    _Float16* p = (_Float16*)(base + off);
    off = (off + n * 2 + 255) & ~(size_t)255;
    return p;
  };

  float* xmel = allocF((size_t)BATCH * 16 * T0);
  float* h1   = allocF((size_t)BATCH * C1 * T1);
  float* gnst = allocF((size_t)BATCH * 2);
  float* h2b  = allocF((size_t)BATCH * C2 * T2);
  float* x    = allocF((size_t)BATCH * SEQ * DIM);

  const int Mrows = BATCH * SEQ;
  _Float16* xn_h   = allocH((size_t)Mrows * DIM);
  _Float16* qkv_h  = allocH((size_t)Mrows * 3 * DIM);
  _Float16* ao_h   = allocH((size_t)Mrows * DIM);
  _Float16* ffn_h  = allocH((size_t)Mrows * 4 * DIM);
  _Float16* wqkv_h = allocH((size_t)NLAYERS * 3 * DIM * DIM);
  _Float16* wo_h   = allocH((size_t)NLAYERS * DIM * DIM);
  _Float16* w1_h   = allocH((size_t)NLAYERS * 4 * DIM * DIM);
  _Float16* w2_h   = allocH((size_t)NLAYERS * DIM * 4 * DIM);

  // weight conversion (once per launch)
  {
    int n1 = NLAYERS * 3 * DIM * DIM;
    k_f32_to_f16<<<(n1 + 255) / 256, 256, 0, stream>>>(tl_Wqkv, wqkv_h, n1);
    int n2 = NLAYERS * DIM * DIM;
    k_f32_to_f16<<<(n2 + 255) / 256, 256, 0, stream>>>(tl_Wo, wo_h, n2);
    int n3 = NLAYERS * 4 * DIM * DIM;
    k_f32_to_f16<<<(n3 + 255) / 256, 256, 0, stream>>>(tl_W1, w1_h, n3);
    k_f32_to_f16<<<(n3 + 255) / 256, 256, 0, stream>>>(tl_W2, w2_h, n3);
  }

  int n;
  n = BATCH * 16 * T0;
  k_mel<<<(n + 255) / 256, 256, 0, stream>>>(mel, mel_W, mel_b, xmel);
  n = BATCH * C1 * T1;
  k_conv1<<<(n + 255) / 256, 256, 0, stream>>>(xmel, conv1_w, conv1_b, h1);
  k_gn_stats<<<BATCH, 256, 0, stream>>>(h1, gnst);
  k_gn_apply<<<(n + 255) / 256, 256, 0, stream>>>(h1, gnst, gn_g, gn_b);
  n = BATCH * C2 * T2;
  k_conv2<<<(n + 255) / 256, 256, 0, stream>>>(h1, conv2_w, conv2_b, h2b);
  k_ln_pos_star<<<BATCH * SEQ, 256, 0, stream>>>(h2b, cn_g, cn_b, star_table,
                                                 star, x);
  k_event_embed<<<BATCH * NEVT, 256, 0, stream>>>(events, emask, ep_W1, ep_b1,
                                                  ep_W2, ep_b2, x);

  dim3 blk(256);
  for (int l = 0; l < NLAYERS; ++l) {
    k_layernorm<<<Mrows, 256, 0, stream>>>(x, tl_ln1g + l * DIM,
                                           tl_ln1b + l * DIM, xn_h);
    {  // QKV projection -> f16
      dim3 g((3 * DIM) / GBN, Mrows / GBM);
      k_gemm_wmma<<<g, blk, 0, stream>>>(xn_h, wqkv_h + (size_t)l * 3 * DIM * DIM,
          tl_bqkv + l * 3 * DIM, nullptr, qkv_h,
          Mrows, 3 * DIM, DIM, DIM, DIM, 3 * DIM, 0, 1, 1.0f);
    }
    {
      dim3 g((SEQ + 15) / 16, BATCH * HEADS);
      k_attention<<<g, blk, 0, stream>>>(qkv_h, ao_h);
    }
    {  // out-projection + residual -> f32 x
      dim3 g(DIM / GBN, Mrows / GBM);
      k_gemm_wmma<<<g, blk, 0, stream>>>(ao_h, wo_h + (size_t)l * DIM * DIM,
          tl_bo + l * DIM, x, x,
          Mrows, DIM, DIM, DIM, DIM, DIM, 0, 0, 1.0f);
    }
    k_layernorm<<<Mrows, 256, 0, stream>>>(x, tl_ln2g + l * DIM,
                                           tl_ln2b + l * DIM, xn_h);
    {  // FFN1 + gelu -> f16
      dim3 g((4 * DIM) / GBN, Mrows / GBM);
      k_gemm_wmma<<<g, blk, 0, stream>>>(xn_h, w1_h + (size_t)l * 4 * DIM * DIM,
          tl_b1 + l * 4 * DIM, nullptr, ffn_h,
          Mrows, 4 * DIM, DIM, DIM, DIM, 4 * DIM, 1, 1, 1.0f);
    }
    {  // FFN2 + residual -> f32 x
      dim3 g(DIM / GBN, Mrows / GBM);
      k_gemm_wmma<<<g, blk, 0, stream>>>(ffn_h, w2_h + (size_t)l * DIM * 4 * DIM,
          tl_b2 + l * DIM, x, x,
          Mrows, DIM, 4 * DIM, 4 * DIM, 4 * DIM, DIM, 0, 0, 1.0f);
    }
  }
  k_pool_head<<<BATCH, 256, 0, stream>>>(x, pool_q, on_g, on_b, oh_W, oh_b,
                                         (float*)d_out);
}
